// RNNetwork_44805098832005
// MI455X (gfx1250) — matrix operand order repaired and checked by reading, not compiled, not run on hardware
//
#include <hip/hip_runtime.h>
#include <hip/hip_bf16.h>

// ---------------------------------------------------------------------------
// CDNA5 (gfx1250) particle-filter battery SOC kernel.
// Dominant cost: per-step GEMM [1024x1024] @ [1024x512] -> v_wmma_f32_16x16x32_f16
// ---------------------------------------------------------------------------

typedef __attribute__((ext_vector_type(16))) _Float16 half16;
typedef __attribute__((ext_vector_type(8)))  float    v8f;
typedef __attribute__((ext_vector_type(4)))  float    f32x4;

#define NP       1024
#define T_STEPS  128
#define HID1     1024
#define HID2     512

#define CUR_MIN  (-2.0f)
#define CUR_MAX  (4.0f)
#define E_CRIT   (29000.0f)
#define F_STD    (0.005f)
#define G_STD    (0.01f)
#define LOG_NU   (3.6862316f)      // log(1/(G_STD*sqrt(2*pi)))
#define V_L      (-1.59614486f)
#define V_0      (4.13646328f)
#define GAMMA_C  (0.63726463f)
#define ALPHA_C  (1.40174122f)
#define BETA_C   (2.54478965f)

__device__ __forceinline__ float sigmoidf(float x) {
    return 1.0f / (1.0f + __expf(-x));
}

__device__ __forceinline__ float voc_f(float s) {
    float sm1 = s - 1.0f;
    return V_L + (V_0 - V_L) * __expf(GAMMA_C * sm1)
         + ALPHA_C * V_L * sm1
         + (1.0f - ALPHA_C) * V_L * (__expf(-BETA_C) - __expf(-BETA_C * sqrtf(s)));
}

// ---------------------------------------------------------------------------
// Pre-swizzle W2 (f32 row-major [1024][512]) into f16 CDNA5 B-fragment order:
//   dst[((kt*32 + ntg)*32 + lane)*16 + e]  <-  W2[k][n]
//   k = 32*kt + 8*(lane>>4) + (e&7) + 16*(e>>3) ,  n = 16*ntg + (lane&15)
// so each lane's 16-element B fragment is one contiguous 32B chunk.
// ---------------------------------------------------------------------------
__global__ void w2_swizzle_kernel(const float* __restrict__ W2,
                                  _Float16* __restrict__ W2h) {
    int tid  = blockIdx.x * 256 + threadIdx.x;    // 0 .. 524287
    int e    = tid & 15;
    int lane = (tid >> 4) & 31;
    int ntg  = (tid >> 9) & 31;
    int kt   = tid >> 14;
    int k = kt * 32 + ((lane >> 4) << 3) + (e & 7) + ((e >> 3) << 4);
    int n = (ntg << 4) + (lane & 15);
    W2h[tid] = (_Float16)W2[k * HID2 + n];
}

__global__ void init_kernel(const float* __restrict__ soc_init,
                            float* __restrict__ soc_state,
                            float* __restrict__ mse) {
    int i = blockIdx.x * 256 + threadIdx.x;
    if (i < NP)      soc_state[i] = soc_init[i];
    if (i < T_STEPS) mse[i] = 0.0f;
}

// ---------------------------------------------------------------------------
// Fused MLP + per-mode epilogue. 64 blocks x 256 threads; 16 particles/block.
// mode 0: prediction (updates soc_state with voc/noise/clip)
// mode 1: measurement (writes logW)
// mode 2: post-resample eval (writes voltage/soc_hist, accumulates mse,
//         commits resampled soc into soc_state)
// ---------------------------------------------------------------------------
__global__ __launch_bounds__(256)
void pf_mlp_kernel(int mode, int t, int cidx,
                   const float* __restrict__ soc_in,
                   const float* __restrict__ cur,
                   const float* __restrict__ vmeas,
                   const float* __restrict__ W1, const float* __restrict__ b1,
                   const _Float16* __restrict__ W2h, const float* __restrict__ b2,
                   const float* __restrict__ W3, const float* __restrict__ b3,
                   const float* __restrict__ noise,
                   float* __restrict__ soc_state, float* __restrict__ logW,
                   float* __restrict__ out_v, float* __restrict__ out_s,
                   float* __restrict__ mse) {
    __shared__ _Float16 h1s[16][HID1];   // 32 KB: layer-1 activations (f16)
    __shared__ float    socL[16];
    __shared__ float    zs[16];

    const int tid   = threadIdx.x;
    const int pbase = blockIdx.x * 16;
    const float c   = cur[cidx];
    const float sI  = (c - CUR_MIN) * (1.0f / (CUR_MAX - CUR_MIN));

    if (tid < 16) { socL[tid] = soc_in[pbase + tid]; zs[tid] = 0.0f; }
    __syncthreads();

    // ---- layer 1: h1 = sigmoid(soc*W1[0,:] + sI*W1[1,:] + b1), rank-2 ----
    #pragma unroll
    for (int jj = 0; jj < 4; ++jj) {
        int j = tid + (jj << 8);
        float w0 = W1[j], w1 = W1[HID1 + j], bb = b1[j];
        float base = sI * w1 + bb;
        #pragma unroll
        for (int p = 0; p < 16; ++p)
            h1s[p][j] = (_Float16)sigmoidf(socL[p] * w0 + base);
    }
    __syncthreads();

    // ---- layer 2: h2 = sigmoid(h1 @ W2 + b2) via WMMA f16 16x16x32 ----
    const int wv   = tid >> 5;     // wave 0..7, owns N columns [64*wv, 64*wv+64)
    const int lane = tid & 31;
    const int row  = lane & 15;    // A-matrix M row for this lane
    const int g    = lane >> 4;    // lane half-group

    v8f acc[4];
    #pragma unroll
    for (int nt = 0; nt < 4; ++nt) acc[nt] = (v8f){};

    const _Float16* h1row = &h1s[row][g << 3];
    const half16* bbase = (const half16*)W2h + ((wv << 2) * 32 + lane);

    for (int kt = 0; kt < 32; ++kt) {
        // A fragment: k = 32*kt + 8*g + {0..7, 16..23}  (ISA 16-bit A layout)
        half16 a;
        const f32x4* ap = (const f32x4*)(h1row + (kt << 5));
        ((f32x4*)&a)[0] = ap[0];     // k offsets +0..7
        ((f32x4*)&a)[1] = ap[2];     // k offsets +16..23
        const half16* bp = bbase + kt * 1024;   // (kt*32 tiles)*32 lanes
        half16 bf0 = bp[0];
        half16 bf1 = bp[32];
        half16 bf2 = bp[64];
        half16 bf3 = bp[96];
        if (kt < 31) __builtin_prefetch((const void*)(bp + 1024), 0, 1);
        acc[0] = __builtin_amdgcn_wmma_f32_16x16x32_f16(false, a, false, bf0,
                                                        (short)0, acc[0], false, false);
        acc[1] = __builtin_amdgcn_wmma_f32_16x16x32_f16(false, a, false, bf1,
                                                        (short)0, acc[1], false, false);
        acc[2] = __builtin_amdgcn_wmma_f32_16x16x32_f16(false, a, false, bf2,
                                                        (short)0, acc[2], false, false);
        acc[3] = __builtin_amdgcn_wmma_f32_16x16x32_f16(false, a, false, bf3,
                                                        (short)0, acc[3], false, false);
    }

    // ---- layer 3: z[p] = sum_col sigmoid(acc + b2[col]) * W3[col] ----
    // C/D layout: element r of v8f -> (row = r + 8*g, col = base + (lane&15))
    float zp[8] = {0, 0, 0, 0, 0, 0, 0, 0};
    #pragma unroll
    for (int nt = 0; nt < 4; ++nt) {
        int col = (wv << 6) + (nt << 4) + (lane & 15);
        float w3 = W3[col], bb = b2[col];
        #pragma unroll
        for (int r = 0; r < 8; ++r)
            zp[r] += sigmoidf(acc[nt][r] + bb) * w3;
    }
    #pragma unroll
    for (int r = 0; r < 8; ++r)
        atomicAdd(&zs[(g << 3) + r], zp[r]);
    __syncthreads();

    // ---- per-particle epilogue ----
    if (tid < 16) {
        int gp   = pbase + tid;
        float soc = socL[tid];
        float z   = zs[tid] + b3[0];
        float V   = voc_f(soc) - c * z;
        if (mode == 0) {
            float eps = noise[t * NP + gp];
            float s2 = soc - c * V * (1.0f / E_CRIT) + eps * F_STD;
            s2 = fminf(fmaxf(s2, 1e-10f), 1.0f);
            soc_state[gp] = s2;
        } else if (mode == 1) {
            float d = (V - vmeas[t]) * (1.0f / G_STD);
            logW[gp] = LOG_NU - 0.5f * d * d;
        } else {
            float vm = vmeas[t];
            out_v[gp * T_STEPS + t] = V;
            out_s[gp * T_STEPS + t] = soc;
            float d = V - vm;
            atomicAdd(&mse[t], d * d);
            soc_state[gp] = soc;   // commit resampled state for next step
        }
    }
}

// ---------------------------------------------------------------------------
// Systematic resampling: single 1024-thread workgroup.
// max-reduce -> exp weights -> Hillis-Steele inclusive scan -> lower_bound.
// ---------------------------------------------------------------------------
__global__ __launch_bounds__(1024)
void resample_kernel(int t, const float* __restrict__ logW,
                     const float* __restrict__ u,
                     const float* __restrict__ soc_state,
                     float* __restrict__ soc_rs) {
    __shared__ float sA[NP];
    __shared__ float sB[NP];
    int tid = threadIdx.x;
    float lw = logW[tid];
    sA[tid] = lw;
    __syncthreads();
    for (int s = 512; s > 0; s >>= 1) {
        if (tid < s) sA[tid] = fmaxf(sA[tid], sA[tid + s]);
        __syncthreads();
    }
    float m = sA[0];
    __syncthreads();
    float w = __expf(lw - m);
    sA[tid] = w;
    __syncthreads();
    float* src = sA;
    float* dst = sB;
    for (int off = 1; off < NP; off <<= 1) {
        float v = src[tid];
        if (tid >= off) v += src[tid - off];
        dst[tid] = v;
        __syncthreads();
        float* tmp = src; src = dst; dst = tmp;
    }
    float total = src[NP - 1];
    // searchsorted(cdf, pos) with cdf = cumsum(w/total) == first i : cumw[i] >= pos*total
    float pos = (u[t] + (float)tid) * (1.0f / (float)NP) * total;
    int lo = 0, hi = NP;
    while (lo < hi) {
        int mid = (lo + hi) >> 1;
        if (src[mid] < pos) lo = mid + 1; else hi = mid;
    }
    int idx = lo < NP - 1 ? lo : NP - 1;
    soc_rs[tid] = soc_state[idx];
}

__global__ void loss_kernel(const float* __restrict__ mse, float* __restrict__ out) {
    __shared__ float s[T_STEPS];
    int tid = threadIdx.x;
    s[tid] = mse[tid];
    __syncthreads();
    for (int k = 64; k > 0; k >>= 1) {
        if (tid < k) s[tid] += s[tid + k];
        __syncthreads();
    }
    if (tid == 0) out[0] = s[0] * (1.0f / ((float)T_STEPS * (float)NP));
}

// ---------------------------------------------------------------------------
extern "C" void kernel_launch(void* const* d_in, const int* in_sizes, int n_in,
                              void* d_out, int out_size, void* d_ws, size_t ws_size,
                              hipStream_t stream) {
    const float* soc_init = (const float*)d_in[0];
    const float* current  = (const float*)d_in[1];   // [128]
    const float* vmeas    = (const float*)d_in[2];   // [128]
    const float* W1 = (const float*)d_in[3];         // [2*1024]
    const float* b1 = (const float*)d_in[4];         // [1024]
    const float* W2 = (const float*)d_in[5];         // [1024*512]
    const float* b2 = (const float*)d_in[6];         // [512]
    const float* W3 = (const float*)d_in[7];         // [512]
    const float* b3 = (const float*)d_in[8];         // [1]
    const float* noise = (const float*)d_in[9];      // [128*1024]
    const float* u  = (const float*)d_in[10];        // [128]

    char* ws = (char*)d_ws;
    _Float16* W2h   = (_Float16*)ws;  ws += (size_t)HID1 * HID2 * sizeof(_Float16); // 1 MB
    float* soc_state = (float*)ws;    ws += NP * sizeof(float);
    float* soc_rs    = (float*)ws;    ws += NP * sizeof(float);
    float* logW      = (float*)ws;    ws += NP * sizeof(float);
    float* mse       = (float*)ws;    ws += T_STEPS * sizeof(float);

    float* out      = (float*)d_out;            // [1 + 1024*128 + 1024*128]
    float* out_loss = out;
    float* out_v    = out + 1;
    float* out_s    = out + 1 + (size_t)NP * T_STEPS;

    // One-time-per-call prep: W2 -> f16 B-fragment layout; reset state/accums.
    w2_swizzle_kernel<<<2048, 256, 0, stream>>>(W2, W2h);
    init_kernel<<<4, 256, 0, stream>>>(soc_init, soc_state, mse);

    for (int t = 0; t < T_STEPS; ++t) {
        int cprev = (t == 0) ? 0 : t - 1;
        // prediction with previous current
        pf_mlp_kernel<<<64, 256, 0, stream>>>(0, t, cprev, soc_state, current, vmeas,
                                              W1, b1, W2h, b2, W3, b3, noise,
                                              soc_state, logW, out_v, out_s, mse);
        // measurement update with current at t
        pf_mlp_kernel<<<64, 256, 0, stream>>>(1, t, t, soc_state, current, vmeas,
                                              W1, b1, W2h, b2, W3, b3, noise,
                                              soc_state, logW, out_v, out_s, mse);
        // systematic resampling
        resample_kernel<<<1, 1024, 0, stream>>>(t, logW, u, soc_state, soc_rs);
        // post-resample evaluation
        pf_mlp_kernel<<<64, 256, 0, stream>>>(2, t, t, soc_rs, current, vmeas,
                                              W1, b1, W2h, b2, W3, b3, noise,
                                              soc_state, logW, out_v, out_s, mse);
    }
    loss_kernel<<<1, 128, 0, stream>>>(mse, out_loss);
}
